// Encoder_Model_3607772528918
// MI455X (gfx1250) — compile-verified, matrix operand order checked by reading, not run on hardware
//
#include <hip/hip_runtime.h>
#include <hip/hip_bf16.h>
#include <stdint.h>

// ---------------------------------------------------------------------------
// GRU encoder, CDNA5 (gfx1250) wave32 + v_wmma_f32_16x16x32_bf16.
//  - Wx/Wh converted+packed once into exact B-fragment lane order (bf16).
//  - embedding converted once to bf16 [VOCAB,128] (zero-padded K).
//  - input projection x_t @ Wx fused into every GRU step (no 550MB xp buffer).
//  - recurrent state h kept in f32 (accuracy) + bf16 shadow (WMMA A operand).
// ---------------------------------------------------------------------------

typedef __attribute__((ext_vector_type(16))) __bf16 v16bf;
typedef __attribute__((ext_vector_type(8)))  float  v8f;

#define B_SZ   256
#define T_SZ   256
#define EMB    100
#define EMBP   128      // EMB padded to 4 K-tiles of 32
#define DY     200
#define DZ     500
#define H      700
#define HP     704      // H padded to 44 N-tiles / 22 K-tiles
#define G3     2100     // 3*H
#define NT     44       // HP/16
#define KT_H   22       // HP/32
#define KT_X   4        // EMBP/32
#define VOCAB  30000

__device__ __forceinline__ uint16_t f32_to_bf16(float f) {
  union { float f; uint32_t u; } c; c.f = f;
  uint32_t r = (c.u + 0x7FFFu + ((c.u >> 16) & 1u)) >> 16;   // RNE
  return (uint16_t)r;
}

union FragU { uint4 q[2]; v16bf v; };

// A fragment (16x32 bf16) from a row-major bf16 matrix.
// Per ISA: lane L holds row M=L%16; K runs [kb, kb+8) and [kb+16, kb+24),
// kb = ktile*32 + (L/16)*8.  `p` already points at row + kb.
__device__ __forceinline__ v16bf load_frag_A(const uint16_t* __restrict__ p) {
  FragU f;
  f.q[0] = *(const uint4*)(p);        // K = kb .. kb+7
  f.q[1] = *(const uint4*)(p + 16);   // K = kb+16 .. kb+23
  return f.v;
}

// B fragment (32x16 bf16) from the pre-packed array: 512 bf16 per tile,
// lane L's 16 elements contiguous at tile*512 + L*16.
__device__ __forceinline__ v16bf load_frag_B(const uint16_t* __restrict__ p) {
  FragU f;
  f.q[0] = *(const uint4*)(p);
  f.q[1] = *(const uint4*)(p + 8);
  return f.v;
}

__device__ __forceinline__ v8f wmma_bf16(v16bf a, v16bf b, v8f c) {
  // emits v_wmma_f32_16x16x32_bf16
  return __builtin_amdgcn_wmma_f32_16x16x32_bf16(false, a, false, b,
                                                 (short)0, c, false, false);
}

// ---------------------------------------------------------------------------
// one-time preprocessing kernels
// ---------------------------------------------------------------------------

__global__ void k_emb_pack(const float* __restrict__ emb,
                           uint16_t* __restrict__ out) {
  int i = blockIdx.x * blockDim.x + threadIdx.x;
  if (i >= VOCAB * EMBP) return;
  int r = i / EMBP, c = i % EMBP;
  float v = (c < EMB) ? emb[(size_t)r * EMB + c] : 0.f;
  out[i] = f32_to_bf16(v);
}

// Pack W [K, 3H] (row-major f32) into bf16 B-fragment order:
// out[g][nt][kt][512], element e = lane*16 + j; (lane,j) -> (k_local,n_local)
// mirrors the ISA 16-bit fragment layout.
__global__ void k_w_pack(const float* __restrict__ W, uint16_t* __restrict__ out,
                         int kt_count, int K_valid) {
  int i = blockIdx.x * blockDim.x + threadIdx.x;
  int total = 3 * NT * kt_count * 512;
  if (i >= total) return;
  int e = i & 511;
  int tile = i >> 9;
  int kt = tile % kt_count; tile /= kt_count;
  int nt = tile % NT;
  int g  = tile / NT;
  int lane = e >> 4, j = e & 15;
  int v = j >> 1, half = j & 1;
  int k_local = ((v & 4) ? 16 : 0) + (lane >> 4) * 8 + (v & 3) * 2 + half;
  int n_local = lane & 15;
  int K   = kt * 32 + k_local;
  int col = nt * 16 + n_local;
  float val = 0.f;
  if (K < K_valid && col < H) val = W[(size_t)K * G3 + g * H + col];
  out[i] = f32_to_bf16(val);
}

__global__ void k_h0(const float* __restrict__ labels, const float* __restrict__ W1,
                     const float* __restrict__ b1, float* __restrict__ hF,
                     uint16_t* __restrict__ hB) {
  int i = blockIdx.x * blockDim.x + threadIdx.x;
  if (i >= B_SZ * HP) return;
  int b = i / HP, c = i % HP;
  float v = 0.f;
  if (c < DY) v = labels[b] * W1[c] + b1[c];
  hF[i] = v;
  hB[i] = f32_to_bf16(v);
}

// ---------------------------------------------------------------------------
// one GRU time step (fused input projection + recurrence + gates)
// grid: 88 blocks x 256 threads = 704 waves; wave -> one 16x16 h-output tile
// ---------------------------------------------------------------------------

__global__ void __launch_bounds__(256) k_gru_step(
    const uint16_t* __restrict__ embB, const uint16_t* __restrict__ wxP,
    const uint16_t* __restrict__ whP,  const int* __restrict__ tokens,
    const float* __restrict__ bias,    const float* __restrict__ hF_in,
    const uint16_t* __restrict__ hB_in, float* __restrict__ hF_out,
    uint16_t* __restrict__ hB_out, int t) {
  const int lane = threadIdx.x & 31;
  const int wave = threadIdx.x >> 5;
  const int idx  = blockIdx.x * 8 + wave;         // 0..703
  const int mt   = idx / NT, nt = idx % NT;
  const int row    = mt * 16 + (lane & 15);       // A-operand row for this lane
  const int kshift = (lane >> 4) * 8;

  v8f accZ = {}, accR = {}, accHX = {}, accHR = {};

  // ---- input projection: A rows gathered from embedding via token ids ----
  const int tok = tokens[(size_t)row * T_SZ + t];
  const uint16_t* arow = embB + (size_t)tok * EMBP + kshift;
#pragma unroll
  for (int kt = 0; kt < KT_X; ++kt) {
    v16bf a = load_frag_A(arow + kt * 32);
    const uint16_t* bz = wxP + ((size_t)((0 * NT + nt) * KT_X + kt)) * 512 + lane * 16;
    const uint16_t* br = wxP + ((size_t)((1 * NT + nt) * KT_X + kt)) * 512 + lane * 16;
    const uint16_t* bh = wxP + ((size_t)((2 * NT + nt) * KT_X + kt)) * 512 + lane * 16;
    accZ  = wmma_bf16(a, load_frag_B(bz), accZ);
    accR  = wmma_bf16(a, load_frag_B(br), accR);
    accHX = wmma_bf16(a, load_frag_B(bh), accHX);
  }

  // ---- recurrent projection: h(t) @ Wh ----
  const uint16_t* hrow = hB_in + (size_t)row * HP + kshift;
  for (int kt = 0; kt < KT_H; ++kt) {
    v16bf a = load_frag_A(hrow + kt * 32);
    const uint16_t* bz = whP + ((size_t)((0 * NT + nt) * KT_H + kt)) * 512 + lane * 16;
    const uint16_t* br = whP + ((size_t)((1 * NT + nt) * KT_H + kt)) * 512 + lane * 16;
    const uint16_t* bh = whP + ((size_t)((2 * NT + nt) * KT_H + kt)) * 512 + lane * 16;
    __builtin_prefetch(bz + 512, 0, 1);           // global_prefetch_b8, next kt
    accZ  = wmma_bf16(a, load_frag_B(bz), accZ);
    accR  = wmma_bf16(a, load_frag_B(br), accR);
    accHR = wmma_bf16(a, load_frag_B(bh), accHR);
  }

  // ---- gates + state update (reset_after=True semantics) ----
  const int col = nt * 16 + (lane & 15);
  float b0z = 0, b1z = 0, b0r = 0, b1r = 0, b0h = 0, b1h = 0;
  if (col < H) {
    b0z = bias[0 * G3 + 0 * H + col]; b1z = bias[1 * G3 + 0 * H + col];
    b0r = bias[0 * G3 + 1 * H + col]; b1r = bias[1 * G3 + 1 * H + col];
    b0h = bias[0 * G3 + 2 * H + col]; b1h = bias[1 * G3 + 2 * H + col];
  }
#pragma unroll
  for (int v = 0; v < 8; ++v) {
    // C/D layout: lanes 0-15 -> M = v, lanes 16-31 -> M = v + 8
    int m = mt * 16 + v + ((lane >> 4) ? 8 : 0);
    float z  = 1.f / (1.f + __expf(-(accZ[v] + b0z + b1z)));
    float r  = 1.f / (1.f + __expf(-(accR[v] + b0r + b1r)));
    float hh = tanhf(accHX[v] + b0h + r * (accHR[v] + b1h));
    float hold = hF_in[(size_t)m * HP + col];
    float hn = z * hold + (1.f - z) * hh;
    if (col >= H) hn = 0.f;                       // keep zero-padded K columns
    hF_out[(size_t)m * HP + col] = hn;
    hB_out[(size_t)m * HP + col] = f32_to_bf16(hn);
  }
}

__global__ void k_out(const float* __restrict__ hF, float* __restrict__ out) {
  int i = blockIdx.x * blockDim.x + threadIdx.x;
  if (i >= B_SZ * DZ) return;
  int b = i / DZ, j = i % DZ;
  out[i] = hF[(size_t)b * HP + DY + j];
}

// ---------------------------------------------------------------------------
// launcher
// ---------------------------------------------------------------------------

extern "C" void kernel_launch(void* const* d_in, const int* in_sizes, int n_in,
                              void* d_out, int out_size, void* d_ws, size_t ws_size,
                              hipStream_t stream) {
  (void)in_sizes; (void)n_in; (void)out_size; (void)ws_size;
  const int*   enc       = (const int*)  d_in[0];  // [B,T]
  const float* labels    = (const float*)d_in[1];  // [B]
  const float* embedding = (const float*)d_in[2];  // [VOCAB,EMB]
  const float* W1        = (const float*)d_in[3];  // [1,DY]
  const float* b1        = (const float*)d_in[4];  // [DY]
  const float* Wx        = (const float*)d_in[5];  // [EMB,3H]
  const float* Wh        = (const float*)d_in[6];  // [H,3H]
  const float* bias      = (const float*)d_in[7];  // [2,3H]
  float* out = (float*)d_out;

  // workspace carve-up (total ~13.4 MB)
  char* ws = (char*)d_ws;
  size_t off = 0;
  auto carve = [&](size_t bytes) {
    char* p = ws + off;
    off = (off + bytes + 255) & ~(size_t)255;
    return p;
  };
  uint16_t* embB = (uint16_t*)carve((size_t)VOCAB * EMBP * 2);
  uint16_t* wxP  = (uint16_t*)carve((size_t)3 * NT * KT_X * 512 * 2);
  uint16_t* whP  = (uint16_t*)carve((size_t)3 * NT * KT_H * 512 * 2);
  float*    hF0  = (float*)   carve((size_t)B_SZ * HP * 4);
  float*    hF1  = (float*)   carve((size_t)B_SZ * HP * 4);
  uint16_t* hB0  = (uint16_t*)carve((size_t)B_SZ * HP * 2);
  uint16_t* hB1  = (uint16_t*)carve((size_t)B_SZ * HP * 2);

  // one-time conversions / packs (cheap; weights stay L2-resident afterwards)
  k_emb_pack<<<(VOCAB * EMBP + 255) / 256, 256, 0, stream>>>(embedding, embB);
  k_w_pack<<<(3 * NT * KT_X * 512 + 255) / 256, 256, 0, stream>>>(Wx, wxP, KT_X, EMB);
  k_w_pack<<<(3 * NT * KT_H * 512 + 255) / 256, 256, 0, stream>>>(Wh, whP, KT_H, H);
  k_h0<<<(B_SZ * HP + 255) / 256, 256, 0, stream>>>(labels, W1, b1, hF0, hB0);

  // sequential GRU: 256 step kernels, double-buffered state
  for (int t = 0; t < T_SZ; ++t) {
    const float*    hin  = (t & 1) ? hF1 : hF0;
    const uint16_t* hbin = (t & 1) ? hB1 : hB0;
    float*          hout = (t & 1) ? hF0 : hF1;
    uint16_t*       hbout= (t & 1) ? hB0 : hB1;
    k_gru_step<<<88, 256, 0, stream>>>(embB, wxP, whP, enc, bias,
                                       hin, hbin, hout, hbout, t);
  }
  // after t=255 the state was written into hF0
  k_out<<<(B_SZ * DZ + 255) / 256, 256, 0, stream>>>(hF0, out);
}
